// ExaoneMoe_10651518894357
// MI455X (gfx1250) — compile-verified, matrix operand order checked by reading, not run on hardware
//
#include <hip/hip_runtime.h>

// ---------------- problem constants ----------------
#define TT      2048          // tokens
#define EE      64            // experts
#define KK      6             // top_k
#define HH      2048          // hidden
#define FF      768           // moe intermediate (== shared intermediate)
#define GG      8             // n_group
#define KGG     4             // topk_group
#define CAP     384           // per-expert capacity
#define NB      (CAP / 64)    // row blocks per expert = 6
#define TNB     (TT / 64)     // row blocks for shared expert = 32
#define SCALE_F 2.5f

#define AS      40            // LDS stride (shorts) for A tile rows  (64 x 32 data)
#define BS      40            // LDS stride (shorts) for B tile cols  (128 x 32 data, col-major)
#define ATILE   (64 * AS)
#define BTILE   (128 * BS)

// ---------------- WMMA types ----------------
typedef __attribute__((ext_vector_type(16))) __bf16 bf16x16;
typedef __attribute__((ext_vector_type(8)))  float  f32x8;

union FragU { bf16x16 v; uint4 q[2]; };

__device__ __forceinline__ unsigned short f2bf(float f) {
  unsigned int u = __float_as_uint(f);
  u = (u + 0x7FFFu + ((u >> 16) & 1u)) >> 16;
  return (unsigned short)u;
}

// A fragment: 16x32 bf16 (M x K) from row-major LDS tile [64][AS].
// lane L: M = m0 + L%16 ; needs K=[khi,khi+8) and [16+khi,16+khi+8), khi=(L>=16)*8
// -> two contiguous 16B chunks -> 2x ds_load_b128.
__device__ __forceinline__ bf16x16 load_frag_A(const unsigned short* sA, int m0, int lane) {
  const unsigned short* rp = sA + (m0 + (lane & 15)) * AS + ((lane >> 4) << 3);
  FragU f;
  f.q[0] = *(const uint4*)(rp);
  f.q[1] = *(const uint4*)(rp + 16);
  return f.v;
}

// B fragment: 32x16 bf16 (K x N) from column-major LDS tile [128][BS].
__device__ __forceinline__ bf16x16 load_frag_B(const unsigned short* sB, int n0, int lane) {
  const unsigned short* cp = sB + (n0 + (lane & 15)) * BS + ((lane >> 4) << 3);
  FragU f;
  f.q[0] = *(const uint4*)(cp);
  f.q[1] = *(const uint4*)(cp + 16);
  return f.v;
}

// ---- software-pipelined B tile staging: 32(K) x 128(N) fp32 tile -> regs -> bf16 LDS
struct Bst { float4 r[4]; };   // 2 units x (row k, row k+1)

__device__ __forceinline__ Bst fetch_B(const float* __restrict__ W, int k0, int n0,
                                       int ldW, int kmax, int tid) {
  Bst s;
#pragma unroll
  for (int i = 0; i < 2; ++i) {
    int u  = tid + i * 256;
    int kp = u & 15;          // k pair: rows 2kp, 2kp+1
    int cg = u >> 4;          // column group of 4
    const float* p0 = W + (size_t)(k0 + 2 * kp) * ldW + n0 + cg * 4;
    s.r[2 * i]     = *(const float4*)(p0);
    s.r[2 * i + 1] = *(const float4*)(p0 + ldW);
    if (k0 + 32 < kmax) __builtin_prefetch(p0 + (size_t)32 * ldW, 0, 0);
  }
  return s;
}

__device__ __forceinline__ void store_B(const Bst& s, unsigned short* sB, int tid) {
#pragma unroll
  for (int i = 0; i < 2; ++i) {
    int u  = tid + i * 256;
    int kp = u & 15;
    int cg = u >> 4;
    float4 r0 = s.r[2 * i], r1 = s.r[2 * i + 1];
    unsigned d0 = ((unsigned)f2bf(r1.x) << 16) | (unsigned)f2bf(r0.x);
    unsigned d1 = ((unsigned)f2bf(r1.y) << 16) | (unsigned)f2bf(r0.y);
    unsigned d2 = ((unsigned)f2bf(r1.z) << 16) | (unsigned)f2bf(r0.z);
    unsigned d3 = ((unsigned)f2bf(r1.w) << 16) | (unsigned)f2bf(r0.w);
    unsigned short* c = sB + (cg * 4) * BS + 2 * kp;
    *(unsigned*)(c + 0 * BS) = d0;
    *(unsigned*)(c + 1 * BS) = d1;
    *(unsigned*)(c + 2 * BS) = d2;
    *(unsigned*)(c + 3 * BS) = d3;
  }
}

// ---------------- kernel 0: router logits + bf16 copy of x ----------------
__global__ __launch_bounds__(64) void moe_router(const float* __restrict__ x,
                                                 const float* __restrict__ gw,
                                                 float* __restrict__ logits,
                                                 unsigned short* __restrict__ xbf) {
  int t = blockIdx.x;
  int e = threadIdx.x;
  const float* xr = x + (size_t)t * HH;
  float acc = 0.f;
#pragma unroll 4
  for (int h = 0; h < HH; ++h) acc = fmaf(xr[h], gw[h * EE + e], acc);
  logits[t * EE + e] = acc;
  for (int h = e; h < HH; h += 64) xbf[(size_t)t * HH + h] = f2bf(xr[h]);
}

// ---------------- kernel 1: grouped top-k ----------------
__global__ __launch_bounds__(128) void moe_topk(const float* __restrict__ logits,
                                                const float* __restrict__ bias,
                                                int* __restrict__ ids,
                                                float* __restrict__ wts) {
  int t = blockIdx.x * blockDim.x + threadIdx.x;
  if (t >= TT) return;
  float sc[EE], sb[EE];
#pragma unroll
  for (int e = 0; e < EE; ++e) {
    float l = logits[t * EE + e];
    float s = 1.f / (1.f + __expf(-l));
    sc[e] = s;
    sb[e] = s + bias[e];
  }
  float gsc[GG];
#pragma unroll
  for (int g = 0; g < GG; ++g) {
    float m1 = -1e30f, m2 = -1e30f;
#pragma unroll
    for (int j = 0; j < EE / GG; ++j) {
      float v = sb[g * (EE / GG) + j];
      if (v > m1) { m2 = m1; m1 = v; } else if (v > m2) { m2 = v; }
    }
    gsc[g] = m1 + m2;
  }
  unsigned gsel = 0;
#pragma unroll
  for (int i = 0; i < KGG; ++i) {
    float bv = -1e30f; int bg = 0;
#pragma unroll
    for (int g = 0; g < GG; ++g)
      if (!((gsel >> g) & 1u) && gsc[g] > bv) { bv = gsc[g]; bg = g; }
    gsel |= 1u << bg;
  }
  unsigned long long used = 0ull;
  int id6[KK]; float w6[KK]; float wsum = 0.f;
#pragma unroll
  for (int i = 0; i < KK; ++i) {
    float bv = -1e30f; int be = 0;
    for (int e = 0; e < EE; ++e)
      if (((gsel >> (e >> 3)) & 1u) && !((used >> e) & 1ull) && sb[e] > bv) { bv = sb[e]; be = e; }
    used |= 1ull << be;
    id6[i] = be; w6[i] = sc[be]; wsum += sc[be];
  }
  float inv = SCALE_F / wsum;
#pragma unroll
  for (int i = 0; i < KK; ++i) {
    ids[t * KK + i] = id6[i];
    wts[t * KK + i] = w6[i] * inv;
  }
}

// ---------------- kernel 2: stable per-expert token lists (one wave32 per expert) ----
__global__ __launch_bounds__(32) void moe_build_lists(const int* __restrict__ ids,
                                                      const float* __restrict__ tw,
                                                      int* __restrict__ ecount,
                                                      int* __restrict__ etok,
                                                      float* __restrict__ ewt) {
  int e = blockIdx.x;
  int lane = threadIdx.x;
  int cnt = 0;
  for (int base = 0; base < TT * KK; base += 32) {
    int pair = base + lane;
    bool m = (ids[pair] == e);
    unsigned long long mk = __ballot(m);
    int pre = __popcll(mk & ((1ull << lane) - 1ull));
    if (m) {
      int slot = cnt + pre;
      if (slot < CAP) {
        etok[e * CAP + slot] = pair / KK;
        ewt[e * CAP + slot]  = tw[pair];
      }
    }
    cnt += __popcll(mk);
  }
  int c = cnt > CAP ? CAP : cnt;
  if (lane == 0) ecount[e] = c;
  for (int s = c + lane; s < CAP; s += 32) { etok[e * CAP + s] = -1; ewt[e * CAP + s] = 0.f; }
}

// ---------------- kernel 3: gate/up GEMM + SwiGLU (double-buffered WMMA bf16) ------
// grid.x = num_row_blocks (routed: EE*NB, shared: TNB); grid.y = FF/128
__global__ __launch_bounds__(256) void moe_gateup(const unsigned short* __restrict__ xbf,
                                                  const float* __restrict__ Wg,
                                                  const float* __restrict__ Wu,
                                                  const int* __restrict__ tok_list,
                                                  const int* __restrict__ counts,
                                                  unsigned short* __restrict__ hout,
                                                  int nb_per_e) {
  int bx = blockIdx.x;
  int e  = bx / nb_per_e;
  int rb = bx % nb_per_e;
  int n0 = blockIdx.y * 128;
  if (counts && rb * 64 >= counts[e]) return;

  __shared__ unsigned short sA[2 * ATILE];
  __shared__ unsigned short sBg[2 * BTILE];
  __shared__ unsigned short sBu[2 * BTILE];
  __shared__ int sTok[64];

  int tid = threadIdx.x, lane = tid & 31, w = tid >> 5;
  int m0 = (w >> 1) * 16, nb = (w & 1) * 64;

  if (tid < 64) sTok[tid] = tok_list ? tok_list[e * CAP + rb * 64 + tid] : (bx * 64 + tid);
  __syncthreads();

  // loop-invariant A source pointer (branchless: invalid rows read token 0,
  // their garbage output is masked at the down-projection scatter)
  int ar = tid >> 2, ac = (tid & 3) * 8;
  int tok = sTok[ar]; tok = tok < 0 ? 0 : tok;
  const unsigned short* aptr = xbf + (size_t)tok * HH + ac;

  const float* wg = Wg + (size_t)e * HH * FF;
  const float* wu = Wu + (size_t)e * HH * FF;

  f32x8 accg[4] = {}, accu[4] = {};

  // pipeline prologue
  uint4 a_st = *(const uint4*)(aptr);
  Bst   g_st = fetch_B(wg, 0, n0, FF, HH, tid);
  Bst   u_st = fetch_B(wu, 0, n0, FF, HH, tid);
  *(uint4*)(sA + ar * AS + ac) = a_st;
  store_B(g_st, sBg, tid);
  store_B(u_st, sBu, tid);
  __syncthreads();

  int cur = 0;
  for (int k0 = 0; k0 < HH; k0 += 32) {
    bool more = (k0 + 32) < HH;
    if (more) {
      a_st = *(const uint4*)(aptr + k0 + 32);
      g_st = fetch_B(wg, k0 + 32, n0, FF, HH, tid);
      u_st = fetch_B(wu, k0 + 32, n0, FF, HH, tid);
    }
    const unsigned short* cA = sA  + cur * ATILE;
    const unsigned short* cG = sBg + cur * BTILE;
    const unsigned short* cU = sBu + cur * BTILE;

    // stage B fragments two subtiles ahead of their WMMA so the
    // ds_load -> wmma dependency is covered by a full WMMA of latency
    bf16x16 a   = load_frag_A(cA, m0, lane);
    bf16x16 bg0 = load_frag_B(cG, nb + 0,  lane);
    bf16x16 bu0 = load_frag_B(cU, nb + 0,  lane);
    bf16x16 bg1 = load_frag_B(cG, nb + 16, lane);
    bf16x16 bu1 = load_frag_B(cU, nb + 16, lane);

    accg[0] = __builtin_amdgcn_wmma_f32_16x16x32_bf16(false, a, false, bg0, (short)0, accg[0], false, false);
    accu[0] = __builtin_amdgcn_wmma_f32_16x16x32_bf16(false, a, false, bu0, (short)0, accu[0], false, false);
    bg0 = load_frag_B(cG, nb + 32, lane);
    bu0 = load_frag_B(cU, nb + 32, lane);
    accg[1] = __builtin_amdgcn_wmma_f32_16x16x32_bf16(false, a, false, bg1, (short)0, accg[1], false, false);
    accu[1] = __builtin_amdgcn_wmma_f32_16x16x32_bf16(false, a, false, bu1, (short)0, accu[1], false, false);
    bg1 = load_frag_B(cG, nb + 48, lane);
    bu1 = load_frag_B(cU, nb + 48, lane);
    accg[2] = __builtin_amdgcn_wmma_f32_16x16x32_bf16(false, a, false, bg0, (short)0, accg[2], false, false);
    accu[2] = __builtin_amdgcn_wmma_f32_16x16x32_bf16(false, a, false, bu0, (short)0, accu[2], false, false);
    accg[3] = __builtin_amdgcn_wmma_f32_16x16x32_bf16(false, a, false, bg1, (short)0, accg[3], false, false);
    accu[3] = __builtin_amdgcn_wmma_f32_16x16x32_bf16(false, a, false, bu1, (short)0, accu[3], false, false);

    if (more) {
      int nxt = cur ^ 1;
      *(uint4*)(sA + nxt * ATILE + ar * AS + ac) = a_st;
      store_B(g_st, sBg + nxt * BTILE, tid);
      store_B(u_st, sBu + nxt * BTILE, tid);
      __syncthreads();
    }
    cur ^= 1;
  }

  // epilogue: h = silu(g) * u  -> bf16
#pragma unroll
  for (int s = 0; s < 4; ++s) {
#pragma unroll
    for (int r = 0; r < 8; ++r) {
      float g = accg[s][r], u = accu[s][r];
      float h = (g / (1.f + __expf(-g))) * u;
      int row = m0 + r + ((lane >> 4) ? 8 : 0);
      int col = n0 + nb + 16 * s + (lane & 15);
      hout[(size_t)(bx * 64 + row) * FF + col] = f2bf(h);
    }
  }
}

// ---------------- kernel 4: down GEMM + weighted scatter-add -----------------------
// grid.x = num_row_blocks; grid.y = HH/128
__global__ __launch_bounds__(256) void moe_down(const unsigned short* __restrict__ hbuf,
                                                const float* __restrict__ Wd,
                                                const int* __restrict__ tok_list,
                                                const float* __restrict__ wt_list,
                                                const int* __restrict__ counts,
                                                float* __restrict__ out,
                                                int nb_per_e) {
  int bx = blockIdx.x;
  int e  = bx / nb_per_e;
  int rb = bx % nb_per_e;
  int n0 = blockIdx.y * 128;
  if (counts && rb * 64 >= counts[e]) return;

  __shared__ unsigned short sA[2 * ATILE];
  __shared__ unsigned short sB[2 * BTILE];
  __shared__ int   sTok[64];
  __shared__ float sWt[64];

  int tid = threadIdx.x, lane = tid & 31, w = tid >> 5;
  int m0 = (w >> 1) * 16, nb = (w & 1) * 64;

  if (tid < 64) {
    int tok; float wt;
    if (tok_list) { tok = tok_list[e * CAP + rb * 64 + tid]; wt = wt_list[e * CAP + rb * 64 + tid]; }
    else          { tok = bx * 64 + tid;                     wt = 1.f; }
    sTok[tid] = tok; sWt[tid] = wt;
  }
  __syncthreads();

  int ar = tid >> 2, ac = (tid & 3) * 8;
  const unsigned short* aptr = hbuf + (size_t)(bx * 64 + ar) * FF + ac;

  const float* wd = Wd + (size_t)e * FF * HH;
  f32x8 acc[4] = {};

  uint4 a_st = *(const uint4*)(aptr);
  Bst   b_st = fetch_B(wd, 0, n0, HH, FF, tid);
  *(uint4*)(sA + ar * AS + ac) = a_st;
  store_B(b_st, sB, tid);
  __syncthreads();

  int cur = 0;
  for (int k0 = 0; k0 < FF; k0 += 32) {
    bool more = (k0 + 32) < FF;
    if (more) {
      a_st = *(const uint4*)(aptr + k0 + 32);
      b_st = fetch_B(wd, k0 + 32, n0, HH, FF, tid);
    }
    const unsigned short* cA = sA + cur * ATILE;
    const unsigned short* cB = sB + cur * BTILE;

    bf16x16 a  = load_frag_A(cA, m0, lane);
    bf16x16 b0 = load_frag_B(cB, nb + 0,  lane);
    bf16x16 b1 = load_frag_B(cB, nb + 16, lane);

    acc[0] = __builtin_amdgcn_wmma_f32_16x16x32_bf16(false, a, false, b0, (short)0, acc[0], false, false);
    b0 = load_frag_B(cB, nb + 32, lane);
    acc[1] = __builtin_amdgcn_wmma_f32_16x16x32_bf16(false, a, false, b1, (short)0, acc[1], false, false);
    b1 = load_frag_B(cB, nb + 48, lane);
    acc[2] = __builtin_amdgcn_wmma_f32_16x16x32_bf16(false, a, false, b0, (short)0, acc[2], false, false);
    acc[3] = __builtin_amdgcn_wmma_f32_16x16x32_bf16(false, a, false, b1, (short)0, acc[3], false, false);

    if (more) {
      int nxt = cur ^ 1;
      *(uint4*)(sA + nxt * ATILE + ar * AS + ac) = a_st;
      store_B(b_st, sB + nxt * BTILE, tid);
      __syncthreads();
    }
    cur ^= 1;
  }

#pragma unroll
  for (int s = 0; s < 4; ++s) {
#pragma unroll
    for (int r = 0; r < 8; ++r) {
      int row = m0 + r + ((lane >> 4) ? 8 : 0);
      int tok = sTok[row];
      if (tok >= 0) {
        int col = n0 + nb + 16 * s + (lane & 15);
        atomicAdd(out + (size_t)tok * HH + col, acc[s][r] * sWt[row]);
      }
    }
  }
}

// ---------------- host side ----------------
extern "C" void kernel_launch(void* const* d_in, const int* in_sizes, int n_in,
                              void* d_out, int out_size, void* d_ws, size_t ws_size,
                              hipStream_t stream) {
  const float* x       = (const float*)d_in[0];
  const float* gate_w  = (const float*)d_in[1];
  const float* bias    = (const float*)d_in[2];
  const float* w_gate  = (const float*)d_in[3];
  const float* w_up    = (const float*)d_in[4];
  const float* w_down  = (const float*)d_in[5];
  const float* ws_gate = (const float*)d_in[6];
  const float* ws_up   = (const float*)d_in[7];
  const float* ws_down = (const float*)d_in[8];
  float* out = (float*)d_out;

  char* base = (char*)d_ws;
  size_t off = 0;
  auto carve = [&](size_t bytes) -> char* {
    char* p = base + off;
    off = (off + bytes + 255) & ~(size_t)255;
    return p;
  };
  float*          logits  = (float*)carve((size_t)TT * EE * 4);
  int*            tk_ids  = (int*)carve((size_t)TT * KK * 4);
  float*          tk_w    = (float*)carve((size_t)TT * KK * 4);
  int*            ecount  = (int*)carve((size_t)EE * 4);
  int*            etok    = (int*)carve((size_t)EE * CAP * 4);
  float*          ewt     = (float*)carve((size_t)EE * CAP * 4);
  unsigned short* xbf     = (unsigned short*)carve((size_t)TT * HH * 2);
  unsigned short* hrout   = (unsigned short*)carve((size_t)EE * CAP * FF * 2);
  unsigned short* hsh     = (unsigned short*)carve((size_t)TT * FF * 2);
  (void)ws_size; (void)in_sizes; (void)n_in;

  hipMemsetAsync(d_out, 0, (size_t)out_size * sizeof(float), stream);

  moe_router<<<TT, 64, 0, stream>>>(x, gate_w, logits, xbf);
  moe_topk<<<(TT + 127) / 128, 128, 0, stream>>>(logits, bias, tk_ids, tk_w);
  moe_build_lists<<<EE, 32, 0, stream>>>(tk_ids, tk_w, ecount, etok, ewt);

  // routed experts
  dim3 g1(EE * NB, FF / 128);
  moe_gateup<<<g1, 256, 0, stream>>>(xbf, w_gate, w_up, etok, ecount, hrout, NB);
  dim3 g2(EE * NB, HH / 128);
  moe_down<<<g2, 256, 0, stream>>>(hrout, w_down, etok, ewt, ecount, out, NB);

  // shared expert (identity token list, weight 1.0, single "expert")
  dim3 g3(TNB, FF / 128);
  moe_gateup<<<g3, 256, 0, stream>>>(xbf, ws_gate, ws_up, nullptr, nullptr, hsh, TNB);
  dim3 g4(TNB, HH / 128);
  moe_down<<<g4, 256, 0, stream>>>(hsh, ws_down, nullptr, nullptr, nullptr, out, TNB);
}
